// DualBranchModel_69922067578976
// MI455X (gfx1250) — matrix operand order repaired and checked by reading, not compiled, not run on hardware
//
#include <hip/hip_runtime.h>
#include <hip/hip_bf16.h>
#include <math.h>

// ---------------------------------------------------------------------------
// DualBranchModel forward for MI455X (gfx1250, wave32, WMMA).
// All matmuls via v_wmma_f32_16x16x32_f16 (f32 tiles converted to f16 in LDS,
// f32 accumulation). Bias / activation / attention-scale / structure-gate are
// fused into the GEMM epilogue. Deterministic (no atomics).
// Loaders use block-uniform K-tail branching + float4 (b128) global loads +
// packed dword LDS stores to avoid per-element exec-mask serialization.
// ---------------------------------------------------------------------------

typedef __attribute__((ext_vector_type(16))) _Float16 v16h;
typedef __attribute__((ext_vector_type(8)))  float    v8f;

#define Bn   64
#define Nn   400
#define Dn   128
#define Hn   4
#define DHn  32
#define CBn  16          // batch chunk for attention score buffers

#define BM 64
#define BN 64
#define BK 32
#define LDST 34          // f16 LDS row stride (pad vs 64-bank conflicts)

union FragU { unsigned int u[8]; v16h v; };

__device__ __forceinline__ unsigned int pkh2(float x, float y) {
  union { _Float16 h[2]; unsigned int u; } t;
  t.h[0] = (_Float16)x; t.h[1] = (_Float16)y;
  return t.u;
}

// ---------------------------------------------------------------------------
// Generic batched WMMA GEMM:
//   C[z] = act( alpha * A[z] @ B[z] + bias + gate_term )
// z decomposed as b = z/zdiv, h = z%zdiv; per-tensor offsets b*s1 + h*s2.
// transB: B stored as [Ncol][K] (row stride ldb)  -- used for q@k^T.
// gate_sc: adds log(sigmoid(sc*gw+gb)+1e-8) per (row,col) (cross-attention).
// splitk: z selects K-range [z*klen, z*klen+klen); C offset z*sC1 (partials).
// ---------------------------------------------------------------------------
__global__ __launch_bounds__(256) void gemm_wmma(
    const float* __restrict__ A, const float* __restrict__ Bm,
    const float* __restrict__ bias, float* __restrict__ C,
    const float* __restrict__ gate_sc, const float* __restrict__ gw_p,
    const float* __restrict__ gb_p,
    int M, int Ncol, int K, int lda, int ldb, int ldc,
    int zdiv, long long sA1, long long sA2, long long sB1, long long sB2,
    long long sC1, long long sC2, long long sG1,
    float alpha, int act, int transB, int splitk, int klen)
{
  __shared__ __align__(16) _Float16 As[BM * LDST];
  __shared__ __align__(16) _Float16 Bs[BN * LDST];

  const int tid  = threadIdx.x;
  const int wave = tid >> 5, lane = tid & 31;
  const int half = lane >> 4, l16 = lane & 15;
  const int tileM = blockIdx.y * BM, tileN = blockIdx.x * BN;
  const int z = blockIdx.z;

  long long offA, offB, offC, offG = 0;
  int k0, k1;
  if (splitk) {
    k0 = z * klen; k1 = k0 + klen; if (k1 > K) k1 = K;
    offA = 0; offB = 0; offC = (long long)z * sC1;
  } else {
    const int b = z / zdiv, h = z - b * zdiv;
    offA = (long long)b * sA1 + (long long)h * sA2;
    offB = (long long)b * sB1 + (long long)h * sB2;
    offC = (long long)b * sC1 + (long long)h * sC2;
    offG = (long long)b * sG1;
    k0 = 0; k1 = K;
  }
  const float* Ab = A + offA;
  const float* Bb = Bm + offB;
  float* Cb = C + offC;

  const int wr = wave & 3;        // 4 row-groups of 16
  const int wc = wave >> 2;       // 2 col-groups of 32
  v8f c0 = {}; v8f c1 = {};

  // A loader mapping: 64 rows x (4 threads x 8 k)
  const int arow = tid >> 2, acol = (tid & 3) * 8;
  // B !transB mapping: 16 k-pairs x 16 col-quads
  const int bk2 = (tid & 15) * 2, bc4 = (tid >> 4) * 4;

  for (int kk = k0; kk < k1; kk += BK) {
    const bool kfull = (kk + BK <= k1);   // block-uniform

    { // ---- A tile: [64 rows][32 k] f32 -> f16 packed dwords
      const int gr = tileM + arow;
      const float* src = Ab + (long long)gr * lda + kk + acol;
      unsigned int* dst = (unsigned int*)&As[arow * LDST + acol];
      if (kfull) {
        if (gr < M) {
          const float4 a0 = *(const float4*)(src);
          const float4 a1 = *(const float4*)(src + 4);
          dst[0] = pkh2(a0.x, a0.y); dst[1] = pkh2(a0.z, a0.w);
          dst[2] = pkh2(a1.x, a1.y); dst[3] = pkh2(a1.z, a1.w);
        } else {
          dst[0] = 0u; dst[1] = 0u; dst[2] = 0u; dst[3] = 0u;
        }
      } else {
#pragma unroll
        for (int j = 0; j < 8; j += 2) {
          const int gk = kk + acol + j;
          const float x = (gr < M && gk < k1)     ? src[j]     : 0.0f;
          const float y = (gr < M && gk + 1 < k1) ? src[j + 1] : 0.0f;
          dst[j >> 1] = pkh2(x, y);
        }
      }
    }

    if (transB) { // ---- B physical [Ncol][K] -> LDS [col][k] (direct)
      const int bc = tid >> 2, bk = (tid & 3) * 8;
      const int gc = tileN + bc;
      const float* src = Bb + (long long)gc * ldb + kk + bk;
      unsigned int* dst = (unsigned int*)&Bs[bc * LDST + bk];
      if (kfull) {
        if (gc < Ncol) {
          const float4 b0 = *(const float4*)(src);
          const float4 b1 = *(const float4*)(src + 4);
          dst[0] = pkh2(b0.x, b0.y); dst[1] = pkh2(b0.z, b0.w);
          dst[2] = pkh2(b1.x, b1.y); dst[3] = pkh2(b1.z, b1.w);
        } else {
          dst[0] = 0u; dst[1] = 0u; dst[2] = 0u; dst[3] = 0u;
        }
      } else {
#pragma unroll
        for (int j = 0; j < 8; j += 2) {
          const int gk = kk + bk + j;
          const float x = (gc < Ncol && gk < k1)     ? src[j]     : 0.0f;
          const float y = (gc < Ncol && gk + 1 < k1) ? src[j + 1] : 0.0f;
          dst[j >> 1] = pkh2(x, y);
        }
      }
    } else {      // ---- B physical [K][Ncol] -> LDS [col][k] (transpose)
      const int gk = kk + bk2;
      const float* s0 = Bb + (long long)gk * ldb + tileN + bc4;
      const float* s1 = s0 + ldb;
      if (kfull) {
        if (tileN + bc4 + 4 <= Ncol) {
          const float4 r0 = *(const float4*)s0;
          const float4 r1 = *(const float4*)s1;
          *(unsigned int*)&Bs[(bc4 + 0) * LDST + bk2] = pkh2(r0.x, r1.x);
          *(unsigned int*)&Bs[(bc4 + 1) * LDST + bk2] = pkh2(r0.y, r1.y);
          *(unsigned int*)&Bs[(bc4 + 2) * LDST + bk2] = pkh2(r0.z, r1.z);
          *(unsigned int*)&Bs[(bc4 + 3) * LDST + bk2] = pkh2(r0.w, r1.w);
        } else {
#pragma unroll
          for (int j = 0; j < 4; ++j) {
            const int gc = tileN + bc4 + j;
            const float x = (gc < Ncol) ? s0[j] : 0.0f;
            const float y = (gc < Ncol) ? s1[j] : 0.0f;
            *(unsigned int*)&Bs[(bc4 + j) * LDST + bk2] = pkh2(x, y);
          }
        }
      } else {
#pragma unroll
        for (int j = 0; j < 4; ++j) {
          const int gc = tileN + bc4 + j;
          const float x = (gc < Ncol && gk < k1)     ? s0[j] : 0.0f;
          const float y = (gc < Ncol && gk + 1 < k1) ? s1[j] : 0.0f;
          *(unsigned int*)&Bs[(bc4 + j) * LDST + bk2] = pkh2(x, y);
        }
      }
    }
    __syncthreads();

    // A fragment (16x32 f16): lane m = wr*16 + l16, dword g holds K pair:
    //   g<4: K = 2g + 8*half ; g>=4: K = 16 + 2(g-4) + 8*half   (ISA 7.12.2)
    FragU fa, fb0, fb1;
    const int m = wr * 16 + l16;
#pragma unroll
    for (int g = 0; g < 8; ++g) {
      const int ka = (g < 4) ? (2 * g + 8 * half) : (16 + 2 * (g - 4) + 8 * half);
      fa.u[g] = *(const unsigned int*)&As[m * LDST + ka];
    }
    // B fragment (32x16): lane col n = l16, K pair same as A fragment
    const int n0 = wc * 32 + l16;
#pragma unroll
    for (int g = 0; g < 8; ++g) {
      const int kb = (g < 4) ? (2 * g + 8 * half) : (16 + 2 * (g - 4) + 8 * half);
      fb0.u[g] = *(const unsigned int*)&Bs[n0 * LDST + kb];
      fb1.u[g] = *(const unsigned int*)&Bs[(n0 + 16) * LDST + kb];
    }
    c0 = __builtin_amdgcn_wmma_f32_16x16x32_f16(false, fa.v, false, fb0.v,
                                                (short)0, c0, false, false);
    c1 = __builtin_amdgcn_wmma_f32_16x16x32_f16(false, fa.v, false, fb1.v,
                                                (short)0, c1, false, false);
    __syncthreads();
  }

  // Epilogue. C layout: dword j -> row = j + 8*half, col = l16 (ISA 7.12.2).
  float gw = 0.f, gb = 0.f;
  if (gate_sc) { gw = *gw_p; gb = *gb_p; }
#pragma unroll
  for (int t = 0; t < 2; ++t) {
    const v8f acc = t ? c1 : c0;
    const int col = tileN + wc * 32 + t * 16 + l16;
    if (col >= Ncol) continue;
    const float bv = bias ? bias[col] : 0.0f;
#pragma unroll
    for (int j = 0; j < 8; ++j) {
      const int row = tileM + wr * 16 + j + 8 * half;
      if (row >= M) continue;
      float v = acc[j] * alpha + bv;
      if (gate_sc) {
        const float s = gate_sc[offG + (long long)row * Ncol + col];
        const float sig = 1.0f / (1.0f + __expf(-(s * gw + gb)));
        v += __logf(sig + 1e-8f);
      }
      if (act == 1) v = fmaxf(v, 0.0f);
      else if (act == 2) v = 1.0f / (1.0f + __expf(-v));
      Cb[(long long)row * ldc + col] = v;
    }
  }
}

// ---------------- helper kernels (one wave32 per row where applicable) ------

__global__ void k_softmax(float* __restrict__ s, int rows, int len) {
  const int row = blockIdx.x * 8 + (threadIdx.x >> 5);
  const int lane = threadIdx.x & 31;
  if (row >= rows) return;
  float* p = s + (long long)row * len;
  float m = -3.4e38f;
  for (int j = lane; j < len; j += 32) m = fmaxf(m, p[j]);
#pragma unroll
  for (int o = 16; o > 0; o >>= 1) m = fmaxf(m, __shfl_xor(m, o, 32));
  float sum = 0.f;
  for (int j = lane; j < len; j += 32) { float e = __expf(p[j] - m); p[j] = e; sum += e; }
#pragma unroll
  for (int o = 16; o > 0; o >>= 1) sum += __shfl_xor(sum, o, 32);
  const float inv = 1.0f / sum;
  for (int j = lane; j < len; j += 32) p[j] *= inv;
}

// out = LayerNorm(in1 + alpha*in2) * g + b   (D = 128, one wave per row)
__global__ void k_ln(const float* __restrict__ in1, const float* __restrict__ in2,
                     const float* __restrict__ alpha_p,
                     const float* __restrict__ g, const float* __restrict__ b,
                     float* __restrict__ out, int rows)
{
  const int row = blockIdx.x * 8 + (threadIdx.x >> 5);
  const int lane = threadIdx.x & 31;
  if (row >= rows) return;
  const float al = alpha_p ? *alpha_p : 1.0f;
  const float* p1 = in1 + (long long)row * 128;
  const float* p2 = in2 ? in2 + (long long)row * 128 : nullptr;
  float v[4]; float s = 0.f;
#pragma unroll
  for (int j = 0; j < 4; ++j) {
    const int c = lane + 32 * j;
    float x = p1[c];
    if (p2) x += al * p2[c];
    v[j] = x; s += x;
  }
#pragma unroll
  for (int o = 16; o > 0; o >>= 1) s += __shfl_xor(s, o, 32);
  const float mean = s * (1.0f / 128.0f);
  float q = 0.f;
#pragma unroll
  for (int j = 0; j < 4; ++j) { const float d = v[j] - mean; q += d * d; }
#pragma unroll
  for (int o = 16; o > 0; o >>= 1) q += __shfl_xor(q, o, 32);
  const float rs = rsqrtf(q * (1.0f / 128.0f) + 1e-5f);
#pragma unroll
  for (int j = 0; j < 4; ++j) {
    const int c = lane + 32 * j;
    out[(long long)row * 128 + c] = (v[j] - mean) * rs * g[c] + b[c];
  }
}

// dis[b,c] = deg>0 ? rsqrt(deg) : 0, deg = sum_r(sc[r,c]) (+1 if diag==0)
__global__ void k_dis(const float* __restrict__ sc, float* __restrict__ dis) {
  const int i = blockIdx.x * blockDim.x + threadIdx.x;
  if (i >= Bn * Nn) return;
  const int b = i / Nn, c = i - b * Nn;
  const float* scb = sc + (long long)b * Nn * Nn;
  float deg = 0.f;
  for (int r = 0; r < Nn; ++r) deg += scb[r * Nn + c];
  if (scb[c * Nn + c] == 0.0f) deg += 1.0f;
  dis[i] = deg > 0.0f ? rsqrtf(deg) : 0.0f;
}

// AnT[b,r,c] = dis[b,r]*dis[b,c]*(sc[b,c,r] + selfloop)
__global__ void k_ant(const float* __restrict__ sc, const float* __restrict__ dis,
                      float* __restrict__ ant) {
  const long long i = (long long)blockIdx.x * blockDim.x + threadIdx.x;
  if (i >= (long long)Bn * Nn * Nn) return;
  const int c = (int)(i % Nn);
  const long long t = i / Nn;
  const int r = (int)(t % Nn);
  const int b = (int)(t / Nn);
  const float* scb = sc + (long long)b * Nn * Nn;
  float a = scb[(long long)c * Nn + r];
  if (r == c && scb[(long long)r * Nn + r] == 0.0f) a += 1.0f;
  ant[i] = dis[b * Nn + r] * dis[b * Nn + c] * a;
}

__global__ void k_cat(const float* __restrict__ zf, const float* __restrict__ zs,
                      float* __restrict__ out, long long rows) {
  const long long i = (long long)blockIdx.x * blockDim.x + threadIdx.x;
  if (i >= rows * 256) return;
  const long long row = i >> 8; const int c = (int)(i & 255);
  out[i] = (c < 128) ? zf[row * 128 + c] : zs[row * 128 + (c - 128)];
}

__global__ void k_mix(const float* __restrict__ gate, const float* __restrict__ zf,
                      const float* __restrict__ zs, float* __restrict__ out, long long n) {
  const long long i = (long long)blockIdx.x * blockDim.x + threadIdx.x;
  if (i >= n) return;
  const float g = gate[i];
  out[i] = g * zf[i] + (1.0f - g) * zs[i];
}

// reduce split-K partials + bias + relu
__global__ void k_redcls(const float* __restrict__ part, const float* __restrict__ bias,
                         float* __restrict__ out, int nsplit) {
  const int i = blockIdx.x * blockDim.x + threadIdx.x;
  if (i >= Bn * 128) return;
  float a = bias[i & 127];
  for (int s = 0; s < nsplit; ++s) a += part[(long long)s * Bn * 128 + i];
  out[i] = fmaxf(a, 0.0f);
}

// out[64,2] = h[64,128] @ w[128,2] + b
__global__ void k_cls2(const float* __restrict__ h, const float* __restrict__ w,
                       const float* __restrict__ b, float* __restrict__ out) {
  const int i = threadIdx.x;
  if (i >= 128) return;
  const int m = i >> 1, c = i & 1;
  float acc = b[c];
  for (int k = 0; k < 128; ++k) acc += h[m * 128 + k] * w[k * 2 + c];
  out[i] = acc;
}

// ---------------------------------------------------------------------------
extern "C" void kernel_launch(void* const* d_in, const int* in_sizes, int n_in,
                              void* d_out, int out_size, void* d_ws, size_t ws_size,
                              hipStream_t stream)
{
  const long long BND = (long long)Bn * Nn * Dn;     // 3,276,800
  const long long BNN = (long long)Bn * Nn * Nn;     // 10,240,000
  const int Mfull = Bn * Nn;                         // 25600
  const float scale = 0.17677669529663688f;          // DH^-0.5

  if (n_in < 68) return;

  // Workspace layout (floats)
  float* W = (float*)d_ws;
  long long f = 0;
  float* X   = W + f; f += BND;            // FC-branch activations / Z_F / Zf
  float* QKV = W + f; f += 3 * BND;        // qkv buffer, later q/k/v, gate, mix, Zj
  float* T1  = W + f; f += BND;            // attention merged out / CA_F2S / g1 / cls_h
  float* FF  = W + f; f += 4 * BND;        // FF intermediate / GCN temps / CAT / merged O
  float* SC  = W + f; f += BNN;            // attention scores (chunked) ALIAS GCN AnT
  float* DIS = W + f; f += (long long)Bn * Nn;
  float* ZS  = W + f; f += BND;            // Z_S / Zs
  float* CAS = W + f; f += BND;            // CA_S2F
  float* PART= W + f; f += 64LL * Bn * 128;// classifier split-K partials
  if (ws_size < (size_t)f * sizeof(float)) return;

  // ---- input mapping: jax pytree (sorted dict keys) flattening.
  // Fallback: top-level insertion order (fc, sc, params...) detected via sizes.
  int sc_idx, pb;
  if (in_sizes[1] == (int)BNN) { sc_idx = 1; pb = 2; }
  else                         { sc_idx = 67; pb = 1; }
  auto in = [&](int i) { return (const float*)d_in[i]; };
  const float* fcm = in(0);
  const float* scm = in(sc_idx);
  const float* alpha1 = in(pb + 0);
  const float* alpha2 = in(pb + 1);
  // ca dict sorted: bk,bq,bv,gb,gw,ob,ow,wk,wq,wv
  const float *ca_bk[2], *ca_bq[2], *ca_bv[2], *ca_gb[2], *ca_gw[2],
              *ca_ob[2], *ca_ow[2], *ca_wk[2], *ca_wq[2], *ca_wv[2];
  for (int cidx = 0; cidx < 2; ++cidx) {
    const int cb = pb + 2 + 10 * cidx;
    ca_bk[cidx] = in(cb + 0); ca_bq[cidx] = in(cb + 1); ca_bv[cidx] = in(cb + 2);
    ca_gb[cidx] = in(cb + 3); ca_gw[cidx] = in(cb + 4);
    ca_ob[cidx] = in(cb + 5); ca_ow[cidx] = in(cb + 6);
    ca_wk[cidx] = in(cb + 7); ca_wq[cidx] = in(cb + 8); ca_wv[cidx] = in(cb + 9);
  }
  const float* cls1_b = in(pb + 22); const float* cls1_w = in(pb + 23);
  const float* cls2_b = in(pb + 24); const float* cls2_w = in(pb + 25);
  const float* fc_b   = in(pb + 26); const float* fc_w   = in(pb + 27);
  const float* fp_b   = in(pb + 28); const float* fp_w   = in(pb + 29);
  const float* g1_b   = in(pb + 30); const float* g1_w   = in(pb + 31);
  const float* g2_b   = in(pb + 32); const float* g2_w   = in(pb + 33);
  const float* gc1_b  = in(pb + 34); const float* gc1_w  = in(pb + 35);
  const float* gc2_b  = in(pb + 36); const float* gc2_w  = in(pb + 37);
  const float* nf_b   = in(pb + 62); const float* nf_g   = in(pb + 63);
  const float* ns_b   = in(pb + 64); const float* ns_g   = in(pb + 65);

  auto G = [&](const float* A, const float* Bm, const float* bias, float* C,
               const float* gate, const float* gw, const float* gb,
               int M, int Nc, int K, int lda, int ldb, int ldc,
               int nz, int zdiv,
               long long sA1, long long sA2, long long sB1, long long sB2,
               long long sC1, long long sC2, long long sG1,
               float alpha, int act, int transB, int splitk, int klen) {
    dim3 grid((Nc + BN - 1) / BN, (M + BM - 1) / BM, nz);
    gemm_wmma<<<grid, dim3(256), 0, stream>>>(
        A, Bm, bias, C, gate, gw, gb, M, Nc, K, lda, ldb, ldc,
        zdiv, sA1, sA2, sB1, sB2, sC1, sC2, sG1, alpha, act, transB, splitk, klen);
  };
  const float* NP = nullptr;
  const long long sQKV = (long long)Nn * 3 * Dn, sND = (long long)Nn * Dn;
  const long long sHNN = (long long)Hn * Nn * Nn, sNN = (long long)Nn * Nn;

  // ================= FC branch =================
  // x = fc_matrix @ fc_proj_w + b   [25600,400]x[400,128]
  G(fcm, fc_w, fc_b, X, NP, NP, NP, Mfull, Dn, Nn, Nn, Dn, Dn,
    1, 1, 0, 0, 0, 0, 0, 0, 0, 1.f, 0, 0, 0, 0);

  for (int l = 0; l < 2; ++l) {  // layer leaves sorted: ff1_b..qkv_w
    const int lb = pb + 38 + 12 * l;
    const float *ff1_b = in(lb + 0), *ff1_w = in(lb + 1);
    const float *ff2_b = in(lb + 2), *ff2_w = in(lb + 3);
    const float *ln1_b = in(lb + 4), *ln1_g = in(lb + 5);
    const float *ln2_b = in(lb + 6), *ln2_g = in(lb + 7);
    const float *o_b   = in(lb + 8), *o_w   = in(lb + 9);
    const float *qkv_b = in(lb + 10), *qkv_w = in(lb + 11);

    G(X, qkv_w, qkv_b, QKV, NP, NP, NP, Mfull, 3 * Dn, Dn, Dn, 3 * Dn, 3 * Dn,
      1, 1, 0, 0, 0, 0, 0, 0, 0, 1.f, 0, 0, 0, 0);
    for (int c = 0; c < Bn / CBn; ++c) {
      const float* qb = QKV + (long long)c * CBn * sQKV;
      // S = scale * q @ k^T   (k section natural layout == transB layout)
      G(qb, qb + Dn, NP, SC, NP, NP, NP, Nn, Nn, DHn, 3 * Dn, 3 * Dn, Nn,
        CBn * Hn, Hn, sQKV, DHn, sQKV, DHn, sHNN, sNN, 0, scale, 0, 1, 0, 0);
      k_softmax<<<(CBn * Hn * Nn + 7) / 8, 256, 0, stream>>>(SC, CBn * Hn * Nn, Nn);
      // O = P @ v, merged per-head into [B,N,D]
      G(SC, qb + 2 * Dn, NP, T1 + (long long)c * CBn * sND, NP, NP, NP,
        Nn, DHn, Nn, Nn, 3 * Dn, Dn,
        CBn * Hn, Hn, sHNN, sNN, sQKV, DHn, sND, DHn, 0, 1.f, 0, 0, 0, 0);
    }
    G(T1, o_w, o_b, FF, NP, NP, NP, Mfull, Dn, Dn, Dn, Dn, Dn,
      1, 1, 0, 0, 0, 0, 0, 0, 0, 1.f, 0, 0, 0, 0);
    k_ln<<<(Mfull + 7) / 8, 256, 0, stream>>>(X, FF, NP, ln1_g, ln1_b, X, Mfull);
    G(X, ff1_w, ff1_b, FF, NP, NP, NP, Mfull, 4 * Dn, Dn, Dn, 4 * Dn, 4 * Dn,
      1, 1, 0, 0, 0, 0, 0, 0, 0, 1.f, 1, 0, 0, 0);
    G(FF, ff2_w, ff2_b, T1, NP, NP, NP, Mfull, Dn, 4 * Dn, 4 * Dn, Dn, Dn,
      1, 1, 0, 0, 0, 0, 0, 0, 0, 1.f, 0, 0, 0, 0);
    k_ln<<<(Mfull + 7) / 8, 256, 0, stream>>>(X, T1, NP, ln2_g, ln2_b, X, Mfull);
  }  // X == Z_F

  // ================= SC branch (2-layer GCN), AnT aliases SC ==============
  k_dis<<<(Bn * Nn + 255) / 256, 256, 0, stream>>>(scm, DIS);
  k_ant<<<(int)((BNN + 255) / 256), 256, 0, stream>>>(scm, DIS, SC);
  G(scm, gc1_w, NP, FF, NP, NP, NP, Mfull, Dn, Nn, Nn, Dn, Dn,
    1, 1, 0, 0, 0, 0, 0, 0, 0, 1.f, 0, 0, 0, 0);                 // XW
  G(SC, FF, gc1_b, FF + BND, NP, NP, NP, Nn, Dn, Nn, Nn, Dn, Dn, // H1 = relu(AnT@XW+b)
    Bn, 1, sNN, 0, sND, 0, sND, 0, 0, 1.f, 1, 0, 0, 0);
  G(FF + BND, gc2_w, NP, FF + 2 * BND, NP, NP, NP, Mfull, Dn, Dn, Dn, Dn, Dn,
    1, 1, 0, 0, 0, 0, 0, 0, 0, 1.f, 0, 0, 0, 0);                 // HW
  G(SC, FF + 2 * BND, gc2_b, ZS, NP, NP, NP, Nn, Dn, Nn, Nn, Dn, Dn,
    Bn, 1, sNN, 0, sND, 0, sND, 0, 0, 1.f, 0, 0, 0, 0);          // Z_S

  // ================= structure-aware cross attention ======================
  auto cross = [&](const float* Qin, const float* KVin, int ci, float* outbuf) {
    G(Qin,  ca_wq[ci], ca_bq[ci], QKV,            NP, NP, NP, Mfull, Dn, Dn, Dn, Dn, Dn,
      1, 1, 0, 0, 0, 0, 0, 0, 0, 1.f, 0, 0, 0, 0);
    G(KVin, ca_wk[ci], ca_bk[ci], QKV + BND,      NP, NP, NP, Mfull, Dn, Dn, Dn, Dn, Dn,
      1, 1, 0, 0, 0, 0, 0, 0, 0, 1.f, 0, 0, 0, 0);
    G(KVin, ca_wv[ci], ca_bv[ci], QKV + 2 * BND,  NP, NP, NP, Mfull, Dn, Dn, Dn, Dn, Dn,
      1, 1, 0, 0, 0, 0, 0, 0, 0, 1.f, 0, 0, 0, 0);
    for (int c = 0; c < Bn / CBn; ++c) {
      const float* qb  = QKV + (long long)c * CBn * sND;
      const float* kb  = QKV + BND + (long long)c * CBn * sND;
      const float* vb  = QKV + 2 * BND + (long long)c * CBn * sND;
      const float* gsc = scm + (long long)c * CBn * sNN;
      // S = scale*q@k^T + log(sigmoid(sc*gw+gb)+1e-8)  (gate fused in epilogue)
      G(qb, kb, NP, SC, gsc, ca_gw[ci], ca_gb[ci], Nn, Nn, DHn, Dn, Dn, Nn,
        CBn * Hn, Hn, sND, DHn, sND, DHn, sHNN, sNN, sNN, scale, 0, 1, 0, 0);
      k_softmax<<<(CBn * Hn * Nn + 7) / 8, 256, 0, stream>>>(SC, CBn * Hn * Nn, Nn);
      G(SC, vb, NP, FF + (long long)c * CBn * sND, NP, NP, NP,
        Nn, DHn, Nn, Nn, Dn, Dn,
        CBn * Hn, Hn, sHNN, sNN, sND, DHn, sND, DHn, 0, 1.f, 0, 0, 0, 0);
    }
    G(FF, ca_ow[ci], ca_ob[ci], outbuf, NP, NP, NP, Mfull, Dn, Dn, Dn, Dn, Dn,
      1, 1, 0, 0, 0, 0, 0, 0, 0, 1.f, 0, 0, 0, 0);
  };
  cross(X,  ZS, 0, T1);    // CA_F2S
  cross(ZS, X,  1, CAS);   // CA_S2F

  k_ln<<<(Mfull + 7) / 8, 256, 0, stream>>>(X,  T1,  alpha1, nf_g, nf_b, X,  Mfull); // Zf
  k_ln<<<(Mfull + 7) / 8, 256, 0, stream>>>(ZS, CAS, alpha2, ns_g, ns_b, ZS, Mfull); // Zs

  // ================= gated fusion + classifier ============================
  k_cat<<<(int)(((long long)Mfull * 256 + 255) / 256), 256, 0, stream>>>(X, ZS, FF, Mfull);
  G(FF, g1_w, g1_b, T1, NP, NP, NP, Mfull, Dn, 2 * Dn, 2 * Dn, Dn, Dn,
    1, 1, 0, 0, 0, 0, 0, 0, 0, 1.f, 1, 0, 0, 0);
  G(T1, g2_w, g2_b, QKV, NP, NP, NP, Mfull, Dn, Dn, Dn, Dn, Dn,
    1, 1, 0, 0, 0, 0, 0, 0, 0, 1.f, 2, 0, 0, 0);                 // gate (sigmoid)
  k_mix<<<(int)((BND + 255) / 256), 256, 0, stream>>>(QKV, X, ZS, QKV + BND, BND);
  G(QKV + BND, fp_w, fp_b, QKV + 2 * BND, NP, NP, NP, Mfull, Dn, Dn, Dn, Dn, Dn,
    1, 1, 0, 0, 0, 0, 0, 0, 0, 1.f, 0, 0, 0, 0);                 // Zj

  // cls1: [64, 51200] @ [51200, 128], deterministic split-K (64 splits of 800)
  G(QKV + 2 * BND, cls1_w, NP, PART, NP, NP, NP, Bn, 128, Nn * Dn,
    Nn * Dn, 128, 128, 64, 1, 0, 0, 0, 0, (long long)Bn * 128, 0, 0,
    1.f, 0, 0, 1, (Nn * Dn) / 64);
  k_redcls<<<(Bn * 128 + 255) / 256, 256, 0, stream>>>(PART, cls1_b, T1, 64);
  k_cls2<<<1, 128, 0, stream>>>(T1, cls2_w, cls2_b, (float*)d_out);
}
